// TTTFeatureAdapter_78091095376127
// MI455X (gfx1250) — compile-verified, matrix operand order checked by reading, not run on hardware
//
#include <hip/hip_runtime.h>

typedef __attribute__((ext_vector_type(16))) _Float16 v16h;
typedef __attribute__((ext_vector_type(8)))  float    v8f;
typedef unsigned int u32x4 __attribute__((ext_vector_type(4)));
typedef int          i32x8 __attribute__((ext_vector_type(8)));
typedef int          i32x4 __attribute__((ext_vector_type(4)));

#define B_  16
#define T_  2048
#define D_  512
#define BD  64
#define NT  (B_*T_)
#define LN_EPS 1e-6f

#define NBPROJ (16*13*512)   // ksteps * coltiles * (32 lanes * 16 f16)
#define NBOUT  (2*32*512)

#define LDS_ROW 520          // 512 data dwords + 2 * 4 pad dwords (TDM pad)

// ---------------------------------------------------------------------------
// Kernel 0 (one-shot prep): pack [Wk|Wv|Wq|lr_w] and Wo into WMMA B-fragment
// order (ISA 7.12.2: lane -> N, K split by lane-half), converted to f16.
// ---------------------------------------------------------------------------
__global__ __launch_bounds__(256) void ttt_pack(
    const float* __restrict__ Wk, const float* __restrict__ Wv,
    const float* __restrict__ Wq, const float* __restrict__ lrw,
    const float* __restrict__ Wo,
    _Float16* __restrict__ Bp, _Float16* __restrict__ Op)
{
    int gid = blockIdx.x * 256 + threadIdx.x;
    if (gid < NBPROJ) {
        int e    = gid & 15;
        int l    = (gid >> 4) & 31;
        int ctks = gid >> 9;            // ks*13 + ct
        int ct   = ctks % 13;
        int ks   = ctks / 13;
        int kb   = (l >> 4) * 8;
        int kg   = ks * 32 + kb + (e < 8 ? e : e + 8);
        int ncol = ct * 16 + (l & 15);
        float w;
        if      (ncol < 64)   w = Wk[(size_t)kg * BD + ncol];
        else if (ncol < 128)  w = Wv[(size_t)kg * BD + (ncol - 64)];
        else if (ncol < 192)  w = Wq[(size_t)kg * BD + (ncol - 128)];
        else if (ncol == 192) w = lrw[kg];
        else                  w = 0.0f;
        Bp[gid] = (_Float16)w;
    } else {
        int g2 = gid - NBPROJ;
        if (g2 < NBOUT) {
            int e    = g2 & 15;
            int l    = (g2 >> 4) & 31;
            int ctks = g2 >> 9;         // ks*32 + ct
            int ct   = ctks & 31;
            int ks   = ctks >> 5;
            int kb   = (l >> 4) * 8;
            int kg   = ks * 32 + kb + (e < 8 ? e : e + 8);
            int n    = ct * 16 + (l & 15);
            Op[g2] = (_Float16)Wo[(size_t)kg * D_ + n];
        }
    }
}

// ---------------------------------------------------------------------------
// Kernel 1: [K|V|Q|eta] = X @ [Wk|Wv|Wq|lr_w] (+bias, sigmoid on eta col).
// The whole 16x512 f32 X slab is DMA'd to LDS by the Tensor Data Mover
// (one tensor_load_to_lds per workgroup, TENSORcnt-synchronized), with
// TDM padding (4 dwords per 256) so the 520-dword row stride avoids LDS
// bank conflicts. The K-loop is then barrier-free: ds_load_b128 gathers +
// f16 cvt for A, one 32B pre-packed global load for B, one WMMA.
// ---------------------------------------------------------------------------
__global__ __launch_bounds__(416) void ttt_proj(
    const float* __restrict__ X, const _Float16* __restrict__ Bp,
    const float* __restrict__ bk, const float* __restrict__ bv,
    const float* __restrict__ bq, const float* __restrict__ lrb,
    float* __restrict__ Kb, float* __restrict__ Vb, float* __restrict__ Qb,
    float* __restrict__ Eb)
{
    __shared__ __align__(16) float As[16 * LDS_ROW];  // 33280 B
    const int tid  = threadIdx.x;
    const int wave = tid >> 5;
    const int lane = tid & 31;
    const int row0 = blockIdx.x * 16;
    const int m    = lane & 15;
    const int hi   = lane >> 4;
    const int kb   = hi * 8;
    const int ncol = wave * 16 + (lane & 15);

    if (wave == 0) {
        // Tensor DMA descriptor (D#) groups 0/1, 2-D tile: 16 rows x 512 f32
        unsigned long long ga = (unsigned long long)(uintptr_t)(X + (size_t)row0 * D_);
        unsigned lds = (unsigned)(uintptr_t)(void*)As;
        u32x4 g0;
        g0[0] = 1u;                                   // count=1 (valid D#)
        g0[1] = lds;                                  // lds_addr
        g0[2] = (unsigned)ga;                         // global_addr[31:0]
        g0[3] = (unsigned)(ga >> 32) | (2u << 30);    // global_addr[56:32] | type=2
        i32x8 g1;
        g1[0] = (int)((2u << 16)      // data_size = 4 bytes
                    | (1u << 20)      // pad_enable
                    | (7u << 22)      // pad_interval: every 256 dwords
                    | (3u << 25));    // pad_amount: 4 dwords
        g1[1] = (int)(512u << 16);    // tensor_dim0 = 512 (low 16 bits)
        g1[2] = (int)(((unsigned)NT & 0xffffu) << 16); // dim0 hi=0 | tensor_dim1 lo
        g1[3] = (int)(512u << 16);    // tensor_dim1 hi=0 | tile_dim0 = 512
        g1[4] = 16;                   // tile_dim1 = 16, tile_dim2 = 0
        g1[5] = 512;                  // tensor_dim0_stride = 512
        g1[6] = 0;
        g1[7] = 0;
        i32x4 z4 = {0, 0, 0, 0};
        i32x8 z8 = {0, 0, 0, 0, 0, 0, 0, 0};
        __builtin_amdgcn_tensor_load_to_lds(g0, g1, z4, z4, z8, 0);
        __builtin_amdgcn_s_wait_tensorcnt(0);
    }
    __syncthreads();

    v8f acc = {};
#pragma unroll
    for (int kk = 0; kk < D_; kk += 32) {
        const int pad  = (kk >= 256) ? 4 : 0;         // TDM pad offset, uniform
        const int base = m * LDS_ROW + kk + kb + pad;
        float4 f0 = *(const float4*)(As + base);
        float4 f1 = *(const float4*)(As + base + 4);
        float4 f2 = *(const float4*)(As + base + 16);
        float4 f3 = *(const float4*)(As + base + 20);
        v16h a;
        a[0]  = (_Float16)f0.x; a[1]  = (_Float16)f0.y;
        a[2]  = (_Float16)f0.z; a[3]  = (_Float16)f0.w;
        a[4]  = (_Float16)f1.x; a[5]  = (_Float16)f1.y;
        a[6]  = (_Float16)f1.z; a[7]  = (_Float16)f1.w;
        a[8]  = (_Float16)f2.x; a[9]  = (_Float16)f2.y;
        a[10] = (_Float16)f2.z; a[11] = (_Float16)f2.w;
        a[12] = (_Float16)f3.x; a[13] = (_Float16)f3.y;
        a[14] = (_Float16)f3.z; a[15] = (_Float16)f3.w;
        v16h b = *(const v16h*)(Bp + ((size_t)((kk >> 5) * 13 + wave) << 9) + lane * 16);
        acc = __builtin_amdgcn_wmma_f32_16x16x32_f16(
                  false, a, false, b, (short)0, acc, false, false);
    }

    // C/D layout: VGPR r -> M = r + 8*hi, lane -> N (epilogue runs once)
#pragma unroll
    for (int r = 0; r < 8; ++r) {
        int row   = row0 + r + hi * 8;
        float val = acc[r];
        if      (ncol < 64)   Kb[(size_t)row * BD + ncol]         = val + bk[ncol];
        else if (ncol < 128)  Vb[(size_t)row * BD + (ncol - 64)]  = val + bv[ncol - 64];
        else if (ncol < 192)  Qb[(size_t)row * BD + (ncol - 128)] = val + bq[ncol - 128];
        else if (ncol == 192) Eb[row] = 1.0f / (1.0f + __expf(-(val + lrb[0])));
    }
}

// ---------------------------------------------------------------------------
// Kernel 2: sequential TTT scan. 1 wave per batch element, W in VGPRs
// (2 rows x 64 f32 per lane). k/q via wave-uniform (scalar) loads, LN
// reductions via shfl_xor butterflies, rank-1 update W -= eta * du k^T.
// ---------------------------------------------------------------------------
__device__ __forceinline__ float wsum32(float x) {
#pragma unroll
    for (int s = 16; s; s >>= 1) x += __shfl_xor(x, s, 32);
    return x;
}

__global__ __launch_bounds__(32) void ttt_scan(
    const float* __restrict__ Kb, const float* __restrict__ Vb,
    const float* __restrict__ Qb, const float* __restrict__ Eb,
    const float* __restrict__ W0,
    const float* __restrict__ lng, const float* __restrict__ lnb,
    float* __restrict__ Hb)
{
    const int b  = blockIdx.x;
    const int l  = threadIdx.x;
    const int r0 = 2 * l, r1 = 2 * l + 1;

    float w0[BD], w1[BD];
#pragma unroll
    for (int i = 0; i < BD; ++i) {
        w0[i] = W0[(size_t)r0 * BD + i];
        w1[i] = W0[(size_t)r1 * BD + i];
    }
    const float g0 = lng[r0], g1 = lng[r1];
    const float e0b = lnb[r0], e1b = lnb[r1];
    const float inv = 1.0f / (float)BD;

    for (int t = 0; t < T_; ++t) {
        const size_t bt = (size_t)b * T_ + t;
        const float* kp = Kb + bt * BD;          // uniform -> scalar loads
        const float* qp = Qb + bt * BD;

        if (t + 1 < T_) {                        // hide next token's latency
            __builtin_prefetch(kp + BD, 0, 0);
            __builtin_prefetch(qp + BD, 0, 0);
            __builtin_prefetch(Vb + (bt + 1) * BD, 0, 0);
        }

        const float2 kd = ((const float2*)kp)[l];
        const float2 vd = ((const float2*)(Vb + bt * BD))[l];
        const float2 qd = ((const float2*)qp)[l];
        const float  eta = Eb[bt];

        // u = W k (4 accumulators per row to break FMA chains)
        float a0 = 0, a1 = 0, a2 = 0, a3 = 0, c0 = 0, c1 = 0, c2 = 0, c3 = 0;
#pragma unroll
        for (int i = 0; i < BD; i += 4) {
            float k0 = kp[i], k1 = kp[i + 1], k2 = kp[i + 2], k3 = kp[i + 3];
            a0 = fmaf(k0, w0[i], a0);     a1 = fmaf(k1, w0[i + 1], a1);
            a2 = fmaf(k2, w0[i + 2], a2); a3 = fmaf(k3, w0[i + 3], a3);
            c0 = fmaf(k0, w1[i], c0);     c1 = fmaf(k1, w1[i + 1], c1);
            c2 = fmaf(k2, w1[i + 2], c2); c3 = fmaf(k3, w1[i + 3], c3);
        }
        float u0 = (a0 + a1) + (a2 + a3);
        float u1 = (c0 + c1) + (c2 + c3);

        // LN forward on u
        float S1 = wsum32(u0 + u1);
        float S2 = wsum32(u0 * u0 + u1 * u1);
        float mu  = S1 * inv;
        float var = S2 * inv - mu * mu;
        float rstd = rsqrtf(var + LN_EPS);
        float yh0 = (u0 - mu) * rstd, yh1 = (u1 - mu) * rstd;
        float pred0 = kd.x + yh0 * g0 + e0b;
        float pred1 = kd.y + yh1 * g1 + e1b;

        // LN backward -> du (rank-1 gradient factor)
        float dy0 = (pred0 - vd.x) * (2.0f * inv) * g0;
        float dy1 = (pred1 - vd.y) * (2.0f * inv) * g1;
        float Se  = wsum32(dy0 + dy1);
        float Sey = wsum32(dy0 * yh0 + dy1 * yh1);
        float me = Se * inv, mey = Sey * inv;
        float du0 = rstd * (dy0 - me - yh0 * mey);
        float du1 = rstd * (dy1 - me - yh1 * mey);

        // W -= eta * du ⊗ k
        float s0 = eta * du0, s1 = eta * du1;
#pragma unroll
        for (int i = 0; i < BD; ++i) {
            float ki = kp[i];
            w0[i] = fmaf(-s0, ki, w0[i]);
            w1[i] = fmaf(-s1, ki, w1[i]);
        }

        // hu = W_new q
        a0 = a1 = a2 = a3 = 0; c0 = c1 = c2 = c3 = 0;
#pragma unroll
        for (int i = 0; i < BD; i += 4) {
            float q0 = qp[i], q1 = qp[i + 1], q2 = qp[i + 2], q3 = qp[i + 3];
            a0 = fmaf(q0, w0[i], a0);     a1 = fmaf(q1, w0[i + 1], a1);
            a2 = fmaf(q2, w0[i + 2], a2); a3 = fmaf(q3, w0[i + 3], a3);
            c0 = fmaf(q0, w1[i], c0);     c1 = fmaf(q1, w1[i + 1], c1);
            c2 = fmaf(q2, w1[i + 2], c2); c3 = fmaf(q3, w1[i + 3], c3);
        }
        float hu0 = (a0 + a1) + (a2 + a3);
        float hu1 = (c0 + c1) + (c2 + c3);

        // h = q + LN(hu)
        S1 = wsum32(hu0 + hu1);
        S2 = wsum32(hu0 * hu0 + hu1 * hu1);
        mu = S1 * inv; var = S2 * inv - mu * mu; rstd = rsqrtf(var + LN_EPS);
        float h0 = qd.x + (hu0 - mu) * rstd * g0 + e0b;
        float h1 = qd.y + (hu1 - mu) * rstd * g1 + e1b;
        ((float2*)(Hb + bt * BD))[l] = make_float2(h0, h1);
    }
}

// ---------------------------------------------------------------------------
// Kernel 3: Z = H @ Wo + bo (32768x64 @ 64x512). H slab staged to LDS with
// async load-to-LDS (ASYNCcnt path); B from pre-packed Op fragments.
// ---------------------------------------------------------------------------
__global__ __launch_bounds__(512) void ttt_out(
    const float* __restrict__ Hb, const _Float16* __restrict__ Op,
    const float* __restrict__ bo, float* __restrict__ Z)
{
    __shared__ __align__(16) float Hs[16 * BD];
    const int tid  = threadIdx.x;
    const int wave = tid >> 5;
    const int lane = tid & 31;
    const int row0 = blockIdx.x * 16;
    const int m  = lane & 15;
    const int hi = lane >> 4;
    const int kb = hi * 8;
    const int n1 = wave * 16 + m;
    const int n2 = (wave + 16) * 16 + m;

    if (tid < 256) {
        // 4 KB slab: 16 bytes per thread, global -> LDS without VGPR bounce
        unsigned lds_off = (unsigned)(uintptr_t)(void*)(Hs + tid * 4);
        unsigned long long gaddr =
            (unsigned long long)(uintptr_t)(Hb + (size_t)row0 * BD + tid * 4);
        asm volatile("global_load_async_to_lds_b128 %0, %1, off"
                     :: "v"(lds_off), "v"(gaddr) : "memory");
    }
    asm volatile("s_wait_asynccnt 0" ::: "memory");
    __syncthreads();

    v8f acc1 = {}, acc2 = {};
#pragma unroll
    for (int kk = 0; kk < BD; kk += 32) {
        v16h a;
#pragma unroll
        for (int e = 0; e < 8; ++e) {
            a[e]     = (_Float16)Hs[m * BD + kk + kb + e];
            a[e + 8] = (_Float16)Hs[m * BD + kk + kb + 16 + e];
        }
        int ks = kk >> 5;
        v16h b1 = *(const v16h*)(Op + ((size_t)(ks * 32 + wave)      << 9) + lane * 16);
        v16h b2 = *(const v16h*)(Op + ((size_t)(ks * 32 + wave + 16) << 9) + lane * 16);
        acc1 = __builtin_amdgcn_wmma_f32_16x16x32_f16(
                   false, a, false, b1, (short)0, acc1, false, false);
        acc2 = __builtin_amdgcn_wmma_f32_16x16x32_f16(
                   false, a, false, b2, (short)0, acc2, false, false);
    }
#pragma unroll
    for (int r = 0; r < 8; ++r) {
        int row = row0 + r + hi * 8;
        Z[(size_t)row * D_ + n1] = acc1[r] + bo[n1];
        Z[(size_t)row * D_ + n2] = acc2[r] + bo[n2];
    }
}

// ---------------------------------------------------------------------------
extern "C" void kernel_launch(void* const* d_in, const int* in_sizes, int n_in,
                              void* d_out, int out_size, void* d_ws, size_t ws_size,
                              hipStream_t stream)
{
    (void)in_sizes; (void)n_in; (void)out_size; (void)ws_size;
    const float* x   = (const float*)d_in[0];
    const float* Wk  = (const float*)d_in[1];
    const float* bk  = (const float*)d_in[2];
    const float* Wv  = (const float*)d_in[3];
    const float* bv  = (const float*)d_in[4];
    const float* Wq  = (const float*)d_in[5];
    const float* bq  = (const float*)d_in[6];
    const float* Wo  = (const float*)d_in[7];
    const float* bo  = (const float*)d_in[8];
    const float* lng = (const float*)d_in[9];
    const float* lnb = (const float*)d_in[10];
    const float* lrw = (const float*)d_in[11];
    const float* lrb = (const float*)d_in[12];
    const float* W0  = (const float*)d_in[13];
    float* Z = (float*)d_out;

    // Workspace layout (floats): K,V,Q (NT*64 each), eta (NT), H (NT*64),
    // then f16 fragment packs Bp (208 KB) and Op (64 KB). ~34.0 MB total.
    float* Kb = (float*)d_ws;
    float* Vb = Kb + (size_t)NT * BD;
    float* Qb = Vb + (size_t)NT * BD;
    float* Eb = Qb + (size_t)NT * BD;
    float* Hb = Eb + (size_t)NT;
    _Float16* Bp = (_Float16*)(Hb + (size_t)NT * BD);
    _Float16* Op = Bp + NBPROJ;

    ttt_pack<<<(NBPROJ + NBOUT + 255) / 256, 256, 0, stream>>>(
        Wk, Wv, Wq, lrw, Wo, Bp, Op);
    ttt_proj<<<NT / 16, 416, 0, stream>>>(x, Bp, bk, bv, bq, lrb,
                                          Kb, Vb, Qb, Eb);
    ttt_scan<<<B_, 32, 0, stream>>>(Kb, Vb, Qb, Eb, W0, lng, lnb, Hb);
    ttt_out<<<NT / 16, 512, 0, stream>>>(Hb, Op, bo, Z);
}
